// DisentangledMultiheadAttention_9723805958520
// MI455X (gfx1250) — matrix operand order repaired and checked by reading, not compiled, not run on hardware
//
#include <hip/hip_runtime.h>

// ---------------------------------------------------------------------------
// DeBERTa disentangled attention for MI455X (gfx1250), bf16 WMMA / fp32 accum.
//
// Algebraic optimization: rel_emb[i,j] = rel_tab[j-i+511] (1023 distinct rows
// for S=512), so pos_k/pos_q projections collapse from [S,S,D]@[D,D]
// (137 GFLOP) to [1023,D]@[D,D] (~1 GFLOP), and
//   c2p[b,h,i,j] = q[b,h,i] . pktab[h, j-i+511]
//   p2c[b,h,i,j] = k[b,h,j] . pqtab[h, j-i+511]
// Logits + softmax + attn@V fused in LDS (16x512 fp32 panel per block) to
// avoid materializing the 67MB [B,H,S,S] logits tensor in HBM.
//
// Operands are staged in WMMA fragment order (bf16) in LDS: fragment loads
// are contiguous 32B v16bf reads (ds_load_b128 pairs), staging stores are
// packed 8-byte ds_store_b64 (2x v_cvt_pk_bf16_f32 per float4 group).
// ---------------------------------------------------------------------------

#define Bsz 8
#define Ssz 512
#define Dsz 512
#define Hn  8
#define DH  64
#define NREL 1023   // 2L-1
#define RELPAD 1024

typedef __attribute__((ext_vector_type(16))) __bf16 v16bf;
typedef __attribute__((ext_vector_type(8)))  float  v8f;

__device__ __forceinline__ v8f wmma_bf16(const v16bf& a, const v16bf& b, const v8f& c) {
  return __builtin_amdgcn_wmma_f32_16x16x32_bf16(false, a, false, b, (short)0, c, false, false);
}

__device__ __forceinline__ unsigned pack_bf2(float a, float b) {
  unsigned short ha = __builtin_bit_cast(unsigned short, (__bf16)a);
  unsigned short hb = __builtin_bit_cast(unsigned short, (__bf16)b);
  return (unsigned)ha | ((unsigned)hb << 16);
}
// store 4 floats as 4 consecutive bf16 with one 8-byte LDS store
__device__ __forceinline__ void store_bf4(__bf16* dst, float x, float y, float z, float w) {
  uint2 p; p.x = pack_bf2(x, y); p.y = pack_bf2(z, w);
  *reinterpret_cast<uint2*>(dst) = p;
}

// ISA 7.12.2 fragment layouts (wave32, 16-bit operands, 16x16x32):
//  A (16x32, lane = M + 16*half):  element e -> K = 8*half + (e<8 ? e : e+8)
//  B (32x16, lane = N + 16*half):  element e -> K = 8*half + (e<8 ? e : e+8)
//  C (16x16): vgpr r -> M = r + 8*half, N = lane&15

// ---------------------------------------------------------------------------
// Kernel 1: Y = X[M,512] @ W[512,512] + bias, bf16 WMMA, fp32 accumulate.
// mode 0: Y row-major [M,512]                 (final out projection)
// mode 1: Y scattered to [B,H,S,DH]           (Q/K/V staging)
// mode 2: Y scattered to [H,1024,DH]          (pktab / pqtab, Mvalid=1023)
// Block: 256 threads (8 waves) -> 16 rows x 128 cols; grid (M/16, 4).
// ---------------------------------------------------------------------------
__global__ void __launch_bounds__(256)
gemm512_wmma(const float* __restrict__ X, const float* __restrict__ W,
             const float* __restrict__ bias, float* __restrict__ Y,
             int Mvalid, int mode) {
  // Fragment-order staging buffers (bf16): one 32B row per lane.
  __shared__ __align__(32) __bf16 Asf[32][16];       // shared A frag (1KB)
  __shared__ __align__(32) __bf16 Bsf[8][32][16];    // per-wave B frags (8KB)

  const int tid  = threadIdx.x;
  const int wave = tid >> 5, lane = tid & 31;
  const int half = lane >> 4, l16 = lane & 15;
  const int bm = blockIdx.x, bn = blockIdx.y;

  v8f acc = {};

  for (int kk = 0; kk < Dsz; kk += 32) {
    // Prefetch next W K-tile into cache while staging this one.
    if (kk + 32 < Dsz) {
      int pr = tid >> 3;                 // 32 rows, 8 threads per row
      int pc = (tid & 7) * 16;
      __builtin_prefetch(&W[(size_t)(kk + 32 + pr) * Dsz + bn * 128 + pc], 0, 3);
    }
    // Stage X tile [16x32] -> A-fragment order (128 float4 loads).
    if (tid < 128) {
      int idx = tid * 4;                 // element index = m*32 + k
      int m = idx >> 5, k = idx & 31;    // 4 consecutive k, same m
      int gm = bm * 16 + m;
      float4 v = make_float4(0.f, 0.f, 0.f, 0.f);
      if (gm < Mvalid) v = *(const float4*)&X[(size_t)gm * Dsz + kk + k];
      int hlf = (k >> 3) & 1;            // constant across the 4 elements
      int e0  = (k & 7) + ((k >> 4) << 3);
      store_bf4(&Asf[m + 16 * hlf][e0], v.x, v.y, v.z, v.w);
    }
    // Stage W tile [32x128] -> per-wave B-fragment order.
    // Column-quads: 4 consecutive k, same c -> 4 consecutive fragment
    // elements -> one packed 8B LDS store. Lane-adjacent c keeps the four
    // global dword loads coalesced.
    #pragma unroll
    for (int t = 0; t < 4; ++t) {
      int qid = tid + t * 256;           // 1024 quads
      int c  = qid & 127;
      int k0 = (qid >> 7) * 4;           // 4-aligned -> same 16-block
      const float* src = &W[(size_t)(kk + k0) * Dsz + bn * 128 + c];
      float v0 = src[0 * Dsz], v1 = src[1 * Dsz], v2 = src[2 * Dsz], v3 = src[3 * Dsz];
      int rowk = 16 * ((k0 >> 3) & 1);
      store_bf4(&Bsf[c >> 4][(c & 15) + rowk][(k0 & 7) + ((k0 >> 4) << 3)], v0, v1, v2, v3);
    }
    __syncthreads();

    v16bf a = *(const v16bf*)Asf[lane];          // one contiguous 32B read
    v16bf b = *(const v16bf*)Bsf[wave][lane];    // one contiguous 32B read
    acc = wmma_bf16(a, b, acc);
    __syncthreads();
  }

  #pragma unroll
  for (int r = 0; r < 8; ++r) {
    int gm = bm * 16 + r + 8 * half;
    int gn = bn * 128 + wave * 16 + l16;
    if (gm >= Mvalid) continue;
    float v = acc[r] + bias[gn];
    if (mode == 0) {
      Y[(size_t)gm * Dsz + gn] = v;
    } else if (mode == 1) {            // [B,H,S,DH]; gm = b*S+s, gn = h*DH+dh
      int b = gm >> 9, s = gm & 511, hh = gn >> 6, dh = gn & 63;
      Y[(((size_t)(b * Hn + hh)) * Ssz + s) * DH + dh] = v;
    } else {                           // [H,1024,DH]
      int hh = gn >> 6, dh = gn & 63;
      Y[((size_t)hh * RELPAD + gm) * DH + dh] = v;
    }
  }
}

// ---------------------------------------------------------------------------
// Kernel 2: fused disentangled attention.
// One block = (b, h, 16-query block). 256 threads = 8 waves.
// Phase A: logits[16][512] in LDS = c2c + c2p + p2c (bf16 WMMA, Toeplitz
//          scatter of the positional terms).  Phase B: softmax (scaled by
//          1/sqrt(3*DH)).  Phase C: attn @ V via WMMA, LDS atomic combine.
// ---------------------------------------------------------------------------
__global__ void __launch_bounds__(256)
attn_fused(const float* __restrict__ Q, const float* __restrict__ Kg,
           const float* __restrict__ Vg, const float* __restrict__ PK,
           const float* __restrict__ PQ, float* __restrict__ AO) {
  __shared__ __align__(32) __bf16 qsA[2][32][16];  // q block, A-frag order (2KB)
  __shared__ float lg[16][Ssz];     // logits -> probabilities (32KB)
  __shared__ float red[16][16];     // row reductions
  __shared__ float rowstat[16];     // row max, then row sum
  __shared__ float outs[16][DH];    // attn @ V accumulator

  const int tid  = threadIdx.x;
  const int wave = tid >> 5, lane = tid & 31;
  const int half = lane >> 4, l16 = lane & 15;

  const int ib = blockIdx.x & 31;
  const int h  = (blockIdx.x >> 5) & 7;
  const int b  = blockIdx.x >> 8;
  const int i0 = ib << 4;
  const int bh = (b << 3) + h;

  const float* Qb  = Q  + (size_t)bh * Ssz * DH;
  const float* Kb  = Kg + (size_t)bh * Ssz * DH;
  const float* Vb  = Vg + (size_t)bh * Ssz * DH;
  const float* PKh = PK + (size_t)h * RELPAD * DH;
  const float* PQh = PQ + (size_t)h * RELPAD * DH;

  // Stage q block [16x64] directly in A-fragment order (shared by c2c & c2p
  // across all j-tiles); zero the output accumulator.
  {
    int idx = tid * 4;                     // element = m*64 + d
    int m = idx >> 6, d = idx & 63;        // 4 consecutive d, same m
    float4 v = *(const float4*)&Qb[(size_t)(i0 + m) * DH + d];
    int chunk = d >> 5, k = d & 31;
    int hlf = (k >> 3) & 1;
    int e0  = (k & 7) + ((k >> 4) << 3);
    store_bf4(&qsA[chunk][m + 16 * hlf][e0], v.x, v.y, v.z, v.w);
  }
  for (int idx = tid; idx < 16 * DH; idx += 256) outs[idx >> 6][idx & 63] = 0.0f;
  __syncthreads();

  // ----- Phase A: logits (each wave owns j-tiles jt = wave, wave+8, ...) ---
  for (int jt = wave; jt < Ssz / 16; jt += 8) {
    const int j0 = jt << 4;

    // c2c: q_tile[16x64] x k_tile^T
    v8f acc = {};
    #pragma unroll
    for (int kc = 0; kc < 2; ++kc) {
      v16bf a = *(const v16bf*)qsA[kc][lane];
      v16bf bf;
      const float* src = &Kb[(size_t)(j0 + l16) * DH + kc * 32 + 8 * half];
      #pragma unroll
      for (int e = 0; e < 16; ++e) bf[e] = (__bf16)src[e + (e & 8)];   // two contiguous 32B runs
      acc = wmma_bf16(a, bf, acc);
    }
    #pragma unroll
    for (int r = 0; r < 8; ++r)
      lg[r + 8 * half][j0 + l16] = acc[r];

    // positional terms: r = j - i + 511 spans [rlo, rlo+30] -> two 16-tiles
    const int rlo = j0 - i0 + 496;
    #pragma unroll
    for (int t = 0; t < 2; ++t) {
      const int rt0 = rlo + t * 16;
      const int rn  = rt0 + l16;               // this lane's table row (B N)
      const bool rok = (rn >= 0) && (rn < NREL);

      // c2p: q_i . pktab[j-i+511]
      v8f pacc = {};
      #pragma unroll
      for (int kc = 0; kc < 2; ++kc) {
        v16bf a = *(const v16bf*)qsA[kc][lane];
        v16bf bf;
        const float* src = &PKh[(size_t)rn * DH + kc * 32 + 8 * half];
        #pragma unroll
        for (int e = 0; e < 16; ++e) bf[e] = (__bf16)(rok ? src[e + (e & 8)] : 0.0f);
        pacc = wmma_bf16(a, bf, pacc);
      }
      #pragma unroll
      for (int r = 0; r < 8; ++r) {            // element (m=di, n=rel row)
        int m = r + 8 * half;
        int j = rn + i0 + m - 511;
        if (j >= j0 && j < j0 + 16) lg[m][j] += pacc[r];   // unique (m,j)
      }

      // p2c: k_j . pqtab[j-i+511]
      v8f dacc = {};
      #pragma unroll
      for (int kc = 0; kc < 2; ++kc) {
        v16bf a;
        const float* a0 = &Kb[(size_t)(j0 + l16) * DH + kc * 32 + 8 * half];
        #pragma unroll
        for (int e = 0; e < 8; ++e) {          // two contiguous 32B runs
          a[e]     = (__bf16)a0[e];
          a[e + 8] = (__bf16)a0[e + 16];
        }
        v16bf bf;
        const float* src = &PQh[(size_t)rn * DH + kc * 32 + 8 * half];
        #pragma unroll
        for (int e = 0; e < 16; ++e) bf[e] = (__bf16)(rok ? src[e + (e & 8)] : 0.0f);
        dacc = wmma_bf16(a, bf, dacc);
      }
      #pragma unroll
      for (int r = 0; r < 8; ++r) {            // element (m=jj, n=rel row)
        int m  = r + 8 * half;
        int j  = j0 + m;
        int di = j - rn + 511 - i0;
        if (di >= 0 && di < 16) lg[di][j] += dacc[r];      // unique (di,j)
      }
    }
  }
  __syncthreads();

  // ----- Phase B: softmax over j, scale 1/sqrt(3*DH) -----------------------
  const float scale = 0.072168783649f;   // 1/sqrt(192)
  const int row = tid >> 4, sub = tid & 15;

  float mx = -3.0e38f;
  for (int c = sub; c < Ssz; c += 16) mx = fmaxf(mx, lg[row][c]);
  red[row][sub] = mx;
  __syncthreads();
  if (sub == 0) {
    float m = red[row][0];
    #pragma unroll
    for (int t = 1; t < 16; ++t) m = fmaxf(m, red[row][t]);
    rowstat[row] = m;
  }
  __syncthreads();
  const float rmax = rowstat[row];
  float sm = 0.0f;
  for (int c = sub; c < Ssz; c += 16) {
    float p = __expf((lg[row][c] - rmax) * scale);
    lg[row][c] = p;
    sm += p;
  }
  __syncthreads();
  red[row][sub] = sm;
  __syncthreads();
  if (sub == 0) {
    float s = 0.0f;
    #pragma unroll
    for (int t = 0; t < 16; ++t) s += red[row][t];
    rowstat[row] = s;
  }
  __syncthreads();

  // ----- Phase C: out = P[16x512] @ V[512x64] ------------------------------
  const int nt = wave & 3;          // 16-col tile of DH
  const int kh = wave >> 2;         // K half (0: j<256, 1: j>=256)
  v8f oac = {};
  for (int kk = kh * 256; kk < kh * 256 + 256; kk += 32) {
    v16bf a;
    const float* a0 = &lg[l16][kk + 8 * half];
    #pragma unroll
    for (int e = 0; e < 8; ++e) {              // two contiguous LDS runs
      a[e]     = (__bf16)a0[e];
      a[e + 8] = (__bf16)a0[e + 16];
    }
    v16bf bf;
    const float* vsrc = &Vb[(size_t)(kk + 8 * half) * DH + nt * 16 + l16];
    #pragma unroll
    for (int e = 0; e < 16; ++e) bf[e] = (__bf16)vsrc[(size_t)(e + (e & 8)) * DH];
    oac = wmma_bf16(a, bf, oac);
  }
  #pragma unroll
  for (int r = 0; r < 8; ++r)
    atomicAdd(&outs[r + 8 * half][nt * 16 + l16], oac[r]);
  __syncthreads();

  // normalize and store to [B, S, H*DH] for the output projection
  for (int idx = tid; idx < 16 * DH; idx += 256) {
    int r = idx >> 6, c = idx & 63;
    AO[((size_t)(b * Ssz + i0 + r)) * Dsz + h * DH + c] = outs[r][c] / rowstat[r];
  }
}

// ---------------------------------------------------------------------------
// Host launcher
// ---------------------------------------------------------------------------
extern "C" void kernel_launch(void* const* d_in, const int* in_sizes, int n_in,
                              void* d_out, int out_size, void* d_ws, size_t ws_size,
                              hipStream_t stream) {
  (void)in_sizes; (void)n_in; (void)out_size; (void)ws_size;
  const float* x       = (const float*)d_in[0];
  const float* Wq      = (const float*)d_in[1];
  const float* bq      = (const float*)d_in[2];
  const float* Wk      = (const float*)d_in[3];
  const float* bk      = (const float*)d_in[4];
  const float* Wv      = (const float*)d_in[5];
  const float* bv      = (const float*)d_in[6];
  const float* rel_tab = (const float*)d_in[7];
  const float* Wpk     = (const float*)d_in[8];
  const float* bpk     = (const float*)d_in[9];
  const float* Wpq     = (const float*)d_in[10];
  const float* bpq     = (const float*)d_in[11];
  const float* Wo      = (const float*)d_in[12];
  const float* bo      = (const float*)d_in[13];
  float* out = (float*)d_out;

  const size_t QKV  = (size_t)Bsz * Hn * Ssz * DH;   // 2,097,152 floats
  const size_t TAB  = (size_t)Hn * RELPAD * DH;      //   524,288 floats
  float* ws = (float*)d_ws;
  float* Qd = ws;
  float* Kd = Qd + QKV;
  float* Vd = Kd + QKV;
  float* PKt = Vd + QKV;
  float* PQt = PKt + TAB;
  float* AO = PQt + TAB;                             // [B*S, D]

  const int M = Bsz * Ssz;                           // 4096
  dim3 blk(256);
  dim3 gProj(M / 16, Dsz / 128);                     // (256, 4)
  dim3 gTab(RELPAD / 16, Dsz / 128);                 // (64, 4)

  // content projections -> [B,H,S,DH]
  gemm512_wmma<<<gProj, blk, 0, stream>>>(x, Wq, bq, Qd, M, 1);
  gemm512_wmma<<<gProj, blk, 0, stream>>>(x, Wk, bk, Kd, M, 1);
  gemm512_wmma<<<gProj, blk, 0, stream>>>(x, Wv, bv, Vd, M, 1);
  // relative-position tables -> [H,1024,DH]  (1023 valid rows)
  gemm512_wmma<<<gTab, blk, 0, stream>>>(rel_tab, Wpk, bpk, PKt, NREL, 2);
  gemm512_wmma<<<gTab, blk, 0, stream>>>(rel_tab, Wpq, bpq, PQt, NREL, 2);

  // fused logits + softmax + attn@V : one block per (b, h, 16-query tile)
  attn_fused<<<Bsz * Hn * (Ssz / 16), blk, 0, stream>>>(Qd, Kd, Vd, PKt, PQt, AO);

  // output projection -> d_out [B,S,D]
  gemm512_wmma<<<gProj, blk, 0, stream>>>(AO, Wo, bo, out, M, 0);
}
